// DAM_88519275970682
// MI455X (gfx1250) — matrix operand order repaired and checked by reading, not compiled, not run on hardware
//
#include <hip/hip_runtime.h>

#define NSEQ 256
#define HDIM 512
#define KSET 2048
#define BSZ  256
#define NM1  255

typedef __attribute__((ext_vector_type(16))) _Float16 v16h;
typedef __attribute__((ext_vector_type(8)))  float    v8f;

// A-operand (16x32, M x K, f16), row-major source with leading dim `ld` (halves).
// Order used here: lane<16 -> row=lane, K 0..15; lane>=16 -> row=lane-16, K 16..31 (contiguous 32B).
static __device__ __forceinline__ v16h load_a_16x32(const _Float16* __restrict__ base,
                                                    int ld, int row0, int k0, int lane) {
  const int r    = row0 + (lane & 15);
  const int koff = (lane & 16) ? 16 : 0;
  const _Float16* p = base + (size_t)r * ld + k0 + koff;
  v16h out;
  ((float4*)&out)[0] = *(const float4*)(p);
  ((float4*)&out)[1] = *(const float4*)(p + 8);
  return out;
}

// B-operand (32x16, K x N, f16) from a [col][k] row-major source (leading dim `ld` halves).
// Order used here: lane<16 -> col=lane, K 0..15; lane>=16 -> col=lane-16, K 16..31 (contiguous 32B).
static __device__ __forceinline__ v16h load_b_32x16(const _Float16* __restrict__ base,
                                                    int ld, int col0, int k0, int lane) {
  const int c    = col0 + (lane & 15);
  const int koff = (lane & 16) ? 16 : 0;
  const _Float16* p = base + (size_t)c * ld + k0 + koff;
  v16h out;
  ((float4*)&out)[0] = ((const float4*)p)[0];
  ((float4*)&out)[1] = ((const float4*)p)[1];
  return out;
}

// ---- k1a: row softmax of B_logits (H rows of length N) -> sB (f16, row-major [h][n]) ----
__global__ void k1a_softmaxB(const float* __restrict__ Blog, _Float16* __restrict__ sBh) {
  __shared__ float red[256];
  const int h = blockIdx.x, t = threadIdx.x;
  float x = Blog[h * NSEQ + t];
  red[t] = x; __syncthreads();
  for (int o = 128; o; o >>= 1) { if (t < o) red[t] = fmaxf(red[t], red[t + o]); __syncthreads(); }
  const float mx = red[0]; __syncthreads();
  const float e = __expf(x - mx);
  red[t] = e; __syncthreads();
  for (int o = 128; o; o >>= 1) { if (t < o) red[t] += red[t + o]; __syncthreads(); }
  sBh[h * NSEQ + t] = (_Float16)(e / red[0]);
}

// ---- k1c: f32 -> f16 conversion (dataset, sequences) ----
__global__ void k1c_cvt(const float* __restrict__ s, _Float16* __restrict__ d) {
  const int t = blockIdx.x * 256 + threadIdx.x;
  d[t] = (_Float16)s[t];
}

// ---- k1b: phi[k][h] = sum_n dataset[k][n] * sB[h][n] via WMMA (f16 in, f32 acc) ----
__global__ void k1b_phi(const _Float16* __restrict__ dsh, const _Float16* __restrict__ sBh,
                        _Float16* __restrict__ phi) {
  const int t = threadIdx.x, lane = t & 31, wid = t >> 5;
  const int tile = blockIdx.x * 8 + wid;      // 4096 tiles = 128 kt x 32 ht
  const int kt = tile >> 5, ht = tile & 31;
  v8f acc = {};
  for (int jc = 0; jc < NSEQ / 32; ++jc) {
    v16h a = load_a_16x32(dsh, NSEQ, kt * 16, jc * 32, lane);   // dataset rows k
    v16h b = load_b_32x16(sBh, NSEQ, ht * 16, jc * 32, lane);   // sB rows h as columns
    acc = __builtin_amdgcn_wmma_f32_16x16x32_f16(false, a, false, b, (short)0, acc, false, false);
  }
  const int h = ht * 16 + (lane & 15);
  const int kb = kt * 16 + ((lane & 16) ? 8 : 0);
  for (int r = 0; r < 8; ++r) phi[(size_t)(kb + r) * HDIM + h] = (_Float16)acc[r];
}

// ---- k2: per (h-tile, prefix i): masked softmax of A_logits[i+1] rows, then
//          hat_phi[b, h] = sum_{j<=i} seq[b, j] * A[h, j] via WMMA ----
__global__ void k2_hat(const float* __restrict__ Alog, const _Float16* __restrict__ seqh,
                       _Float16* __restrict__ hat) {
  __shared__ __align__(16) _Float16 Atile[16][NSEQ];   // [local h][j], f16, 8 KB
  const int ht = blockIdx.x;          // 0..31 h-tile
  const int i  = blockIdx.y;          // 0..254 ; prefix len = i+1, row A_logits[i+1]
  const int t = threadIdx.x, lane = t & 31, wid = t >> 5;
  const int rowL = t >> 4, sub = t & 15;   // 16 threads per softmax row

  const float* arow = Alog + ((size_t)(i + 1) * HDIM + ht * 16 + rowL) * NSEQ;
  float e[16];
  float mx = -INFINITY;
  for (int jj = 0; jj < 16; ++jj) { int j = sub + jj * 16; if (j <= i) mx = fmaxf(mx, arow[j]); }
  for (int o = 8; o; o >>= 1) mx = fmaxf(mx, __shfl_xor(mx, o, 16));
  float sum = 0.f;
  for (int jj = 0; jj < 16; ++jj) {
    int j = sub + jj * 16;
    e[jj] = (j <= i) ? __expf(arow[j] - mx) : 0.f;
    sum += e[jj];
  }
  for (int o = 8; o; o >>= 1) sum += __shfl_xor(sum, o, 16);
  const float inv = 1.f / sum;
  for (int jj = 0; jj < 16; ++jj) { int j = sub + jj * 16; Atile[rowL][j] = (_Float16)(e[jj] * inv); }
  __syncthreads();

  const int jch = (i + 32) >> 5;   // ceil((i+1)/32); masked tail is exact zeros
  for (int bt = wid; bt < 16; bt += 8) {
    v8f acc = {};
    for (int jc = 0; jc < jch; ++jc) {
      v16h a = load_a_16x32(seqh, NSEQ, bt * 16, jc * 32, lane);
      v16h b = load_b_32x16(&Atile[0][0], NSEQ, 0, jc * 32, lane);
      acc = __builtin_amdgcn_wmma_f32_16x16x32_f16(false, a, false, b, (short)0, acc, false, false);
    }
    _Float16* hp = hat + (size_t)i * (BSZ * HDIM);
    const int hg = ht * 16 + (lane & 15);
    const int bb = bt * 16 + ((lane & 16) ? 8 : 0);
    for (int r = 0; r < 8; ++r) hp[(size_t)(bb + r) * HDIM + hg] = (_Float16)acc[r];
  }
}

// ---- k3: per prefix i: logits = hat_phi @ phi^T via WMMA (4 k-tiles per softmax
//          update), fused online log-sum-exp + argmax + target logit ----
__global__ void k3_ce(const _Float16* __restrict__ hat, const _Float16* __restrict__ phi,
                      const int* __restrict__ idx, float* __restrict__ ce,
                      float* __restrict__ acw) {
  const int i = blockIdx.x;
  const int t = threadIdx.x, lane = t & 31, wid = t >> 5;
  const _Float16* hp = hat + (size_t)i * (BSZ * HDIM);
  for (int bq = 0; bq < 2; ++bq) {
    const int bt = wid * 2 + bq;                    // b-tile (16 rows)
    const int rbase = bt * 16 + ((lane & 16) ? 8 : 0);
    float m[8], s[8], best[8], tgt[8];
    int bestk[8], tk[8];
    for (int r = 0; r < 8; ++r) {
      m[r] = -INFINITY; s[r] = 0.f; best[r] = -INFINITY; tgt[r] = 0.f;
      bestk[r] = -1; tk[r] = idx[rbase + r];
    }
    for (int kg = 0; kg < KSET / 64; ++kg) {        // 32 groups of 4 k-tiles (64 cols)
      v8f d0 = {}, d1 = {}, d2 = {}, d3 = {};
      for (int hc = 0; hc < HDIM / 32; ++hc) {
        v16h a  = load_a_16x32(hp, HDIM, bt * 16, hc * 32, lane);       // reused 4x
        v16h b0 = load_b_32x16(phi, HDIM, (kg * 4 + 0) * 16, hc * 32, lane);
        d0 = __builtin_amdgcn_wmma_f32_16x16x32_f16(false, a, false, b0, (short)0, d0, false, false);
        v16h b1 = load_b_32x16(phi, HDIM, (kg * 4 + 1) * 16, hc * 32, lane);
        d1 = __builtin_amdgcn_wmma_f32_16x16x32_f16(false, a, false, b1, (short)0, d1, false, false);
        v16h b2 = load_b_32x16(phi, HDIM, (kg * 4 + 2) * 16, hc * 32, lane);
        d2 = __builtin_amdgcn_wmma_f32_16x16x32_f16(false, a, false, b2, (short)0, d2, false, false);
        v16h b3 = load_b_32x16(phi, HDIM, (kg * 4 + 3) * 16, hc * 32, lane);
        d3 = __builtin_amdgcn_wmma_f32_16x16x32_f16(false, a, false, b3, (short)0, d3, false, false);
      }
      for (int r = 0; r < 8; ++r) {
        const float v0 = d0[r], v1 = d1[r], v2 = d2[r], v3 = d3[r];
        // local (per-lane) max over the 4 tiles; strict > keeps lowest k
        float bv = v0; int q = 0;
        if (v1 > bv) { bv = v1; q = 1; }
        if (v2 > bv) { bv = v2; q = 2; }
        if (v3 > bv) { bv = v3; q = 3; }
        int bk = (kg * 4 + q) * 16 + (lane & 15);
        // cross-lane (16 columns) max with first-index tie-break
        for (int o = 8; o; o >>= 1) {
          float ov = __shfl_xor(bv, o, 16);
          int   ok = __shfl_xor(bk, o, 16);
          if (ov > bv || (ov == bv && ok < bk)) { bv = ov; bk = ok; }
        }
        if (bv > best[r]) { best[r] = bv; bestk[r] = bk; }
        // online log-sum-exp over 64 columns at once
        const float mn = fmaxf(m[r], bv);
        float ex = __expf(v0 - mn) + __expf(v1 - mn) + __expf(v2 - mn) + __expf(v3 - mn);
        for (int o = 8; o; o >>= 1) ex += __shfl_xor(ex, o, 16);
        s[r] = s[r] * __expf(m[r] - mn) + ex;
        m[r] = mn;
        // target logit extraction
        const int tq = (tk[r] >> 4) - kg * 4;
        if (tq >= 0 && tq < 4) {
          const float tv = (tq == 0) ? v0 : (tq == 1) ? v1 : (tq == 2) ? v2 : v3;
          tgt[r] = __shfl(tv, (tk[r] & 15) + (lane & 16), 32);
        }
      }
    }
    if ((lane & 15) == 0) {
      for (int r = 0; r < 8; ++r) {
        const int row = rbase + r;
        ce[i * BSZ + row]  = m[r] + __logf(s[r]) - tgt[r];
        acw[i * BSZ + row] = (bestk[r] == tk[r]) ? 1.f : 0.f;
      }
    }
  }
}

// ---- k4: final means -> d_out = {loss, accuracy} ----
__global__ void k4_reduce(const float* __restrict__ ce, const float* __restrict__ acw,
                          float* __restrict__ out) {
  __shared__ float r1[256], r2[256];
  const int t = threadIdx.x;
  float a = 0.f, b = 0.f;
  for (int x = t; x < NM1 * BSZ; x += 256) { a += ce[x]; b += acw[x]; }
  r1[t] = a; r2[t] = b; __syncthreads();
  for (int o = 128; o; o >>= 1) {
    if (t < o) { r1[t] += r1[t + o]; r2[t] += r2[t + o]; }
    __syncthreads();
  }
  if (t == 0) {
    out[0] = r1[0] / (float)(NM1 * BSZ);
    out[1] = r2[0] / (float)(NM1 * BSZ);
  }
}

extern "C" void kernel_launch(void* const* d_in, const int* in_sizes, int n_in,
                              void* d_out, int out_size, void* d_ws, size_t ws_size,
                              hipStream_t stream) {
  (void)in_sizes; (void)n_in; (void)out_size; (void)ws_size;
  const float* Alog    = (const float*)d_in[0];   // (N, H, N)
  const float* Blog    = (const float*)d_in[1];   // (H, N)
  const float* seqf    = (const float*)d_in[2];   // (B, N)
  const float* dataset = (const float*)d_in[3];   // (K, N)
  const int*   indices = (const int*)d_in[4];     // (B,)
  float* out = (float*)d_out;

  char* ws = (char*)d_ws;
  _Float16* sBh  = (_Float16*)(ws);                 // 256 KB  f16 (H,N)
  _Float16* dsh  = (_Float16*)(ws + 262144);        // 1 MB    f16 (K,N)
  _Float16* phi  = (_Float16*)(ws + 1310720);       // 2 MB    f16 (K,H)
  _Float16* seqh = (_Float16*)(ws + 3407872);       // 128 KB  f16 (B,N)
  float*    ce   = (float*)(ws + 3538944);          // 255 KB  f32 (NM1,B)
  float*    acw  = (float*)(ws + 3800064);          // 255 KB  f32 (NM1,B)
  _Float16* hat  = (_Float16*)(ws + 4061184);       // 66.8 MB f16 (NM1,B,H)

  k1a_softmaxB<<<HDIM, 256, 0, stream>>>(Blog, sBh);
  k1c_cvt<<<(KSET * NSEQ) / 256, 256, 0, stream>>>(dataset, dsh);
  k1c_cvt<<<(BSZ * NSEQ) / 256, 256, 0, stream>>>(seqf, seqh);
  k1b_phi<<<512, 256, 0, stream>>>(dsh, sBh, phi);
  k2_hat<<<dim3(32, NM1), 256, 0, stream>>>(Alog, seqh, hat);
  k3_ce<<<NM1, 256, 0, stream>>>(hat, phi, indices, ce, acw);
  k4_reduce<<<1, 256, 0, stream>>>(ce, acw, out);
}
